// SpanPredictor_87024627351863
// MI455X (gfx1250) — matrix operand order repaired and checked by reading, compiled, not run on hardware
//
#include <hip/hip_runtime.h>
#include <hip/hip_bf16.h>

// ---------------------------------------------------------------------------
// SpanPredictor for MI455X (gfx1250, wave32, WMMA).
//
// Algebraic factorization: layer-1 GEMM over the concatenated pair matrix
// decomposes into three shared GEMMs (words@W1b once per WORD instead of once
// per head x position), cutting layer-1 from 215 GFLOP to ~15 GFLOP. All
// GEMMs run on v_wmma_f32_16x16x32_bf16 (fp32 accumulate). The per-head
// pipeline (h1 relu-sum -> GEMM2 -> GEMM3 -> conv1 -> conv2 -> scatter) is
// fused into one kernel that keeps h1 (64x1024 bf16 = 128KB) in CDNA5's
// 320KB LDS.
// ---------------------------------------------------------------------------

typedef __attribute__((ext_vector_type(16))) __bf16 v16bf;
typedef __attribute__((ext_vector_type(8)))  __bf16 v8bf;
typedef __attribute__((ext_vector_type(4)))  __bf16 v4bf;
typedef __attribute__((ext_vector_type(8)))  float  v8f;

#define N_WORDS   8192
#define SENT_LEN  64
#define N_HEADS   1024
#define IN_SZ     768
#define HIDDEN    1024
#define C2DIM     256
#define DIST_EMB  64
#define CONV_CH   4
#define MAX_DIST  128

static __device__ __forceinline__ __bf16 f2bf(float x) {
  unsigned u = __builtin_bit_cast(unsigned, x);
  u += 0x7FFFu + ((u >> 16) & 1u);             // round-to-nearest-even
  unsigned short s = (unsigned short)(u >> 16);
  return __builtin_bit_cast(__bf16, s);
}

// ---- WMMA fragment loaders (contiguous k order in A and B, wave32) -------------
// A (16x32 bf16, row-major source, row stride lda elements):
//   lane m = lane&15, half = lane>>4:
//   elems e = A[m][kb + 16*half + e]
static __device__ __forceinline__ v16bf
load_a_frag(const __bf16* A, int lda, int mbase, int kb, int lane) {
  int m    = mbase + (lane & 15);
  int half = lane >> 4;
  const __bf16* p = A + (size_t)m * lda + kb;
  return *(const v16bf*)(p + 16 * half);
}

// B (32x16 bf16) loaded from a PRE-TRANSPOSED [N,K] matrix:
//   lane n = lane&15, half = lane>>4: elems e -> K = kb + 16*half + e
//   -> one contiguous, 32B-aligned 16-element load.
static __device__ __forceinline__ v16bf
load_b_frag(const __bf16* Bt, int ldb, int nbase, int kb, int lane) {
  int n    = nbase + (lane & 15);
  int half = lane >> 4;
  return *(const v16bf*)(Bt + (size_t)n * ldb + kb + 16 * half);
}

// C/D (16x16 f32): lane n = lane&15, half = lane>>4; VGPR r -> M = r + 8*half.

// ---- prep kernels ----------------------------------------------------------
__global__ void k_cvt_bf16(const float* __restrict__ s, __bf16* __restrict__ d, int n) {
  int i = blockIdx.x * blockDim.x + threadIdx.x;
  if (i < n) d[i] = f2bf(s[i]);
}

// dst[n*K + k] = src[(rowOff + k)*srcLd + n]   (convert + transpose)
__global__ void k_transpose_cvt(const float* __restrict__ src, __bf16* __restrict__ dst,
                                int K, int srcLd, int rowOff, int total) {
  int i = blockIdx.x * blockDim.x + threadIdx.x;
  if (i >= total) return;
  int n = i / K, k = i - n * K;
  dst[(size_t)n * K + k] = f2bf(src[(size_t)(rowOff + k) * srcLd + n]);
}

__global__ void k_gather_heads(const float* __restrict__ words,
                               const int* __restrict__ heads,
                               __bf16* __restrict__ dst) {
  int i = blockIdx.x * blockDim.x + threadIdx.x;
  if (i >= N_HEADS * IN_SZ) return;
  int h = i / IN_SZ, k = i - h * IN_SZ;
  dst[(size_t)h * IN_SZ + k] = f2bf(words[(size_t)heads[h] * IN_SZ + k]);
}

// ---- generic bf16 WMMA GEMM: C[M,N] = A[M,K] (row) * Bt[N,K] (row) --------
// 8 waves/block, block tile 64(M) x 128(N); each wave owns a 16x64 strip.
__global__ __launch_bounds__(256)
void k_gemm_bf16(const __bf16* __restrict__ A, const __bf16* __restrict__ Bt,
                 float* __restrict__ C, int M, int N, int K) {
  int lane = threadIdx.x & 31;
  int wave = threadIdx.x >> 5;           // 0..7
  int wm = wave & 3, wn = wave >> 2;     // 4 x 2 wave grid
  int mbase = blockIdx.y * 64 + wm * 16;
  int nbase = blockIdx.x * 128 + wn * 64;
  v8f acc[4] = {};
  for (int kb = 0; kb < K; kb += 32) {
    v16bf a = load_a_frag(A, K, mbase, kb, lane);
#pragma unroll
    for (int i = 0; i < 4; ++i) {
      v16bf b = load_b_frag(Bt, K, nbase + i * 16, kb, lane);
      acc[i] = __builtin_amdgcn_wmma_f32_16x16x32_bf16(
          false, a, false, b, (short)0, acc[i], false, false);
    }
  }
  int half = lane >> 4, n0 = lane & 15;
#pragma unroll
  for (int i = 0; i < 4; ++i)
#pragma unroll
    for (int r = 0; r < 8; ++r)
      C[(size_t)(mbase + r + 8 * half) * N + nbase + i * 16 + n0] = acc[i][r];
}

// ---- fused per-head pipeline ----------------------------------------------
__global__ __launch_bounds__(256)
void k_fused_head(const float* __restrict__ B_all,    // [8192,1024]
                  const float* __restrict__ A_heads,  // [1024,1024]
                  const float* __restrict__ E1,       // [128,1024]
                  const float* __restrict__ b1,       // [1024]
                  const __bf16* __restrict__ W2t,     // [256,1024]
                  const float* __restrict__ b2,       // [256]
                  const __bf16* __restrict__ W3t,     // [64,256]
                  const float* __restrict__ b3,       // [64]
                  const float* __restrict__ c1w,      // [4,64,3]
                  const float* __restrict__ c1b,      // [4]
                  const float* __restrict__ c2w,      // [2,4,3]
                  const float* __restrict__ c2b,      // [2]
                  const int* __restrict__ sent_id,    // [8192]
                  const int* __restrict__ heads_ids,  // [1024]
                  float* __restrict__ out)            // [1024,8192,2]
{
  extern __shared__ __align__(32) char smem[];
  __bf16* h1  = (__bf16*)smem;                                   // 64*1024 bf16 = 128KB
  __bf16* h2  = (__bf16*)(smem + SENT_LEN * HIDDEN * 2);         // 64*256  bf16 =  32KB
  float*  h3  = (float*)(smem + SENT_LEN * HIDDEN * 2 + SENT_LEN * C2DIM * 2); // 64*64 f32 = 16KB
  float*  c1o = h3 + SENT_LEN * DIST_EMB;                        // 4*64
  float*  res = c1o + CONV_CH * SENT_LEN;                        // 64*2

  const int h    = blockIdx.x;
  const int head = heads_ids[h];
  const int sid  = sent_id[head];
  const int start = sid * SENT_LEN;   // sentences are contiguous uniform runs
  const int tid  = threadIdx.x;
  const int lane = tid & 31;
  const int wave = tid >> 5;

  // ---- stage 1: h1 = relu(A[h] + B_all[row] + E1[id] + b1), store bf16 ----
  for (int idx = tid; idx < SENT_LEN * (HIDDEN / 4); idx += 256) {
    int j  = idx >> 8;               // position in sentence
    int c4 = (idx & 255) * 4;        // column group
    int col = start + j;
    int eid = (head - col) + (MAX_DIST - 2) / 2;
    if (eid < 0 || eid > MAX_DIST - 2) eid = MAX_DIST - 1;
    float4 Bv = *(const float4*)(B_all + (size_t)col * HIDDEN + c4);
    float4 Av = *(const float4*)(A_heads + (size_t)h * HIDDEN + c4);
    float4 Ev = *(const float4*)(E1 + (size_t)eid * HIDDEN + c4);
    float4 bv = *(const float4*)(b1 + c4);
    v4bf o;
    o[0] = f2bf(fmaxf(Bv.x + Av.x + Ev.x + bv.x, 0.f));
    o[1] = f2bf(fmaxf(Bv.y + Av.y + Ev.y + bv.y, 0.f));
    o[2] = f2bf(fmaxf(Bv.z + Av.z + Ev.z + bv.z, 0.f));
    o[3] = f2bf(fmaxf(Bv.w + Av.w + Ev.w + bv.w, 0.f));
    *(v4bf*)(h1 + (size_t)j * HIDDEN + c4) = o;
  }
  __syncthreads();

  // ---- stage 2: h2 = relu(h1 @ W2 + b2)  [64x1024 x 1024x256] -------------
  {
    int wm = wave & 3, wn = wave >> 2;      // 4 x 2
    int mbase = wm * 16;
    int nbase = wn * 128;                   // each wave: 16 x 128 strip
    v8f acc[8] = {};
    for (int kb = 0; kb < HIDDEN; kb += 32) {
      v16bf a = load_a_frag(h1, HIDDEN, mbase, kb, lane);
#pragma unroll
      for (int i = 0; i < 8; ++i) {
        v16bf b = load_b_frag(W2t, HIDDEN, nbase + i * 16, kb, lane);
        acc[i] = __builtin_amdgcn_wmma_f32_16x16x32_bf16(
            false, a, false, b, (short)0, acc[i], false, false);
      }
    }
    int half = lane >> 4, n0 = lane & 15;
#pragma unroll
    for (int i = 0; i < 8; ++i) {
      int col = nbase + i * 16 + n0;
      float bias = b2[col];
#pragma unroll
      for (int r = 0; r < 8; ++r) {
        int row = mbase + r + 8 * half;
        h2[(size_t)row * C2DIM + col] = f2bf(fmaxf(acc[i][r] + bias, 0.f));
      }
    }
  }
  __syncthreads();

  // ---- stage 3: h3 = h2 @ W3 + b3  [64x256 x 256x64] ----------------------
  {
    int wm = wave & 3, wn = wave >> 2;
    int mbase = wm * 16;
    int nbase = wn * 32;                    // each wave: 16 x 32 strip
    v8f acc[2] = {};
    for (int kb = 0; kb < C2DIM; kb += 32) {
      v16bf a = load_a_frag(h2, C2DIM, mbase, kb, lane);
#pragma unroll
      for (int i = 0; i < 2; ++i) {
        v16bf b = load_b_frag(W3t, C2DIM, nbase + i * 16, kb, lane);
        acc[i] = __builtin_amdgcn_wmma_f32_16x16x32_bf16(
            false, a, false, b, (short)0, acc[i], false, false);
      }
    }
    int half = lane >> 4, n0 = lane & 15;
#pragma unroll
    for (int i = 0; i < 2; ++i) {
      int col = nbase + i * 16 + n0;
      float bias = b3[col];
#pragma unroll
      for (int r = 0; r < 8; ++r) {
        int row = mbase + r + 8 * half;
        h3[(size_t)row * DIST_EMB + col] = acc[i][r] + bias;
      }
    }
  }
  __syncthreads();

  // ---- stage 4: conv1 (64ch -> 4ch, k=3, pad 1) over L --------------------
  {
    int co = tid >> 6, l = tid & 63;        // 4*64 = 256 threads exactly
    float acc = c1b[co];
    for (int ci = 0; ci < DIST_EMB; ++ci) {
#pragma unroll
      for (int t = 0; t < 3; ++t) {
        int ll = l - 1 + t;
        if (ll >= 0 && ll < SENT_LEN)
          acc += h3[(size_t)ll * DIST_EMB + ci] * c1w[(co * DIST_EMB + ci) * 3 + t];
      }
    }
    c1o[co * SENT_LEN + l] = acc;
  }
  __syncthreads();

  // ---- conv2 (4ch -> 2ch, k=3, pad 1) -------------------------------------
  if (tid < 2 * SENT_LEN) {
    int co = tid >> 6, l = tid & 63;
    float acc = c2b[co];
#pragma unroll
    for (int ci = 0; ci < CONV_CH; ++ci)
#pragma unroll
      for (int t = 0; t < 3; ++t) {
        int ll = l - 1 + t;
        if (ll >= 0 && ll < SENT_LEN)
          acc += c1o[ci * SENT_LEN + ll] * c2w[(co * CONV_CH + ci) * 3 + t];
      }
    res[l * 2 + co] = acc;
  }
  __syncthreads();

  // ---- stage 5: scatter into -inf-masked [8192,2] row for this head ------
  const float NINF = __int_as_float(0xFF800000);
  float* orow = out + (size_t)h * N_WORDS * 2;
  for (int w = tid; w < N_WORDS; w += 256) {
    float v0 = NINF, v1 = NINF;
    if (w >= start && w < start + SENT_LEN) {
      int j = w - start;
      if (w <= head) v0 = res[j * 2 + 0];   // valid start
      if (w >= head) v1 = res[j * 2 + 1];   // valid end
    }
    *(float2*)(orow + (size_t)w * 2) = make_float2(v0, v1);
  }
}

// ---------------------------------------------------------------------------
extern "C" void kernel_launch(void* const* d_in, const int* in_sizes, int n_in,
                              void* d_out, int out_size, void* d_ws, size_t ws_size,
                              hipStream_t stream) {
  // inputs (setup_inputs dict order)
  const int*   sent_id   = (const int*)  d_in[0];
  const float* words     = (const float*)d_in[1];
  const int*   heads_ids = (const int*)  d_in[2];
  const float* W1        = (const float*)d_in[3];
  const float* b1        = (const float*)d_in[4];
  const float* W2        = (const float*)d_in[5];
  const float* b2        = (const float*)d_in[6];
  const float* W3        = (const float*)d_in[7];
  const float* b3        = (const float*)d_in[8];
  const float* c1w       = (const float*)d_in[9];
  const float* c1b       = (const float*)d_in[10];
  const float* c2w       = (const float*)d_in[11];
  const float* c2b       = (const float*)d_in[12];
  const float* emb       = (const float*)d_in[13];
  float* out = (float*)d_out;

  // workspace carve (all 256B-aligned by construction)
  char* ws = (char*)d_ws;
  size_t off = 0;
  auto carve = [&](size_t bytes) { char* p = ws + off; off += (bytes + 255) & ~size_t(255); return p; };
  __bf16* words_bf = (__bf16*)carve((size_t)N_WORDS * IN_SZ * 2);
  __bf16* headsw   = (__bf16*)carve((size_t)N_HEADS * IN_SZ * 2);
  __bf16* W1a_t    = (__bf16*)carve((size_t)HIDDEN * IN_SZ * 2);
  __bf16* W1b_t    = (__bf16*)carve((size_t)HIDDEN * IN_SZ * 2);
  __bf16* W1c_t    = (__bf16*)carve((size_t)HIDDEN * DIST_EMB * 2);
  __bf16* W2t      = (__bf16*)carve((size_t)C2DIM * HIDDEN * 2);
  __bf16* W3t      = (__bf16*)carve((size_t)DIST_EMB * C2DIM * 2);
  __bf16* emb_bf   = (__bf16*)carve((size_t)MAX_DIST * DIST_EMB * 2);
  float*  B_all    = (float*) carve((size_t)N_WORDS * HIDDEN * 4);
  float*  A_heads  = (float*) carve((size_t)N_HEADS * HIDDEN * 4);
  float*  E1       = (float*) carve((size_t)MAX_DIST * HIDDEN * 4);

  auto blocks = [](int n) { return (n + 255) / 256; };

  // ---- prep: convert / gather / transpose ---------------------------------
  k_cvt_bf16<<<blocks(N_WORDS * IN_SZ), 256, 0, stream>>>(words, words_bf, N_WORDS * IN_SZ);
  k_cvt_bf16<<<blocks(MAX_DIST * DIST_EMB), 256, 0, stream>>>(emb, emb_bf, MAX_DIST * DIST_EMB);
  k_gather_heads<<<blocks(N_HEADS * IN_SZ), 256, 0, stream>>>(words, heads_ids, headsw);
  // W1 is [1600,1024]; split rows [0:768] / [768:1536] / [1536:1600], transpose to [N,K]
  k_transpose_cvt<<<blocks(HIDDEN * IN_SZ), 256, 0, stream>>>(W1, W1a_t, IN_SZ, HIDDEN, 0,    HIDDEN * IN_SZ);
  k_transpose_cvt<<<blocks(HIDDEN * IN_SZ), 256, 0, stream>>>(W1, W1b_t, IN_SZ, HIDDEN, 768,  HIDDEN * IN_SZ);
  k_transpose_cvt<<<blocks(HIDDEN * DIST_EMB), 256, 0, stream>>>(W1, W1c_t, DIST_EMB, HIDDEN, 1536, HIDDEN * DIST_EMB);
  k_transpose_cvt<<<blocks(C2DIM * HIDDEN), 256, 0, stream>>>(W2, W2t, HIDDEN, C2DIM, 0, C2DIM * HIDDEN);
  k_transpose_cvt<<<blocks(DIST_EMB * C2DIM), 256, 0, stream>>>(W3, W3t, C2DIM, DIST_EMB, 0, DIST_EMB * C2DIM);

  // ---- shared layer-1 GEMMs (WMMA bf16) -----------------------------------
  // B_all = words @ W1b : [8192,768]x[768,1024]
  k_gemm_bf16<<<dim3(HIDDEN / 128, N_WORDS / 64), 256, 0, stream>>>(
      words_bf, W1b_t, B_all, N_WORDS, HIDDEN, IN_SZ);
  // A_heads = words[heads] @ W1a : [1024,768]x[768,1024]
  k_gemm_bf16<<<dim3(HIDDEN / 128, N_HEADS / 64), 256, 0, stream>>>(
      headsw, W1a_t, A_heads, N_HEADS, HIDDEN, IN_SZ);
  // E1 = emb @ W1c : [128,64]x[64,1024]
  k_gemm_bf16<<<dim3(HIDDEN / 128, MAX_DIST / 64), 256, 0, stream>>>(
      emb_bf, W1c_t, E1, MAX_DIST, HIDDEN, DIST_EMB);

  // ---- fused per-head pipeline --------------------------------------------
  size_t shmem = (size_t)SENT_LEN * HIDDEN * 2      // h1 bf16  128KB
               + (size_t)SENT_LEN * C2DIM * 2       // h2 bf16   32KB
               + (size_t)SENT_LEN * DIST_EMB * 4    // h3 f32    16KB
               + (size_t)CONV_CH * SENT_LEN * 4     // conv1 out  1KB
               + (size_t)SENT_LEN * 2 * 4;          // res      0.5KB
  k_fused_head<<<N_HEADS, 256, shmem, stream>>>(
      B_all, A_heads, E1, b1, W2t, b2, W3t, b3,
      c1w, c1b, c2w, c2b, sent_id, heads_ids, out);
}
